// VQEmbedding_13786845020515
// MI455X (gfx1250) — matrix operand order repaired and checked by reading, not compiled, not run on hardware
//
#include <hip/hip_runtime.h>
#include <stdint.h>

#define D_DIM   256
#define K_CODES 1024
#define N_TOTAL 65536   // 64 * 32 * 32

typedef __attribute__((ext_vector_type(16))) __bf16 v16bf;
typedef __attribute__((ext_vector_type(8)))  float  v8f;

union Frag32B {
    uint4 u[2];
    v16bf v;
};

__device__ __forceinline__ unsigned short f32_to_bf16_rne(float f) {
    unsigned int u = __float_as_uint(f);
    unsigned int r = u + 0x7fffu + ((u >> 16) & 1u);
    return (unsigned short)(r >> 16);
}

// ---------------------------------------------------------------------------
// Kernel 1: NCHW fp32 -> [N][D] bf16 transpose, LDS-tiled so both the global
// read (contiguous in spatial) and the global write (contiguous in channel)
// are coalesced. grid = (B=64, S/32=32, D/32=8), block = (32, 8).
// ---------------------------------------------------------------------------
__global__ void convert_z_nchw_to_nd_bf16(const float* __restrict__ z,
                                          unsigned short* __restrict__ zb) {
    __shared__ float tile[32][33];  // +1 pad kills bank conflicts
    const int b  = blockIdx.x;          // batch
    const int s0 = blockIdx.y * 32;     // spatial base (h*32+w)
    const int d0 = blockIdx.z * 32;     // channel base
    const int tx = threadIdx.x, ty = threadIdx.y;

#pragma unroll
    for (int j = 0; j < 4; ++j) {
        const int d = d0 + ty + j * 8;
        tile[ty + j * 8][tx] = z[((size_t)(b * 256 + d) << 10) + (s0 + tx)];
    }
    __syncthreads();
#pragma unroll
    for (int j = 0; j < 4; ++j) {
        const int s = s0 + ty + j * 8;
        zb[((size_t)(b * 1024 + s) << 8) + (d0 + tx)] =
            f32_to_bf16_rne(tile[tx][ty + j * 8]);
    }
}

// ---------------------------------------------------------------------------
// Kernel 2: codebook fp32 -> bf16, plus exact fp32 ||e_k||^2 per code.
// One block (256 threads) per code row.
// ---------------------------------------------------------------------------
__global__ void convert_emb_bf16_norms(const float* __restrict__ emb,
                                       unsigned short* __restrict__ eb,
                                       float* __restrict__ enorm) {
    __shared__ float red[256];
    const int k = blockIdx.x;
    const int t = threadIdx.x;
    const float v = emb[k * D_DIM + t];
    eb[k * D_DIM + t] = f32_to_bf16_rne(v);
    red[t] = v * v;
    __syncthreads();
    for (int off = 128; off > 0; off >>= 1) {
        if (t < off) red[t] += red[t + off];
        __syncthreads();
    }
    if (t == 0) enorm[k] = red[0];
}

// ---------------------------------------------------------------------------
// Hot-loop helpers for the fused GEMM+argmin kernel.
// ---------------------------------------------------------------------------

// Issue the 16 global_load_b128 for one 16-code B tile + its ||e||^2 scalar.
// p is the per-lane column base: eb + (cb + lane%16)*D + (lane>=16)*8.
__device__ __forceinline__ void load_btile(Frag32B* __restrict__ b,
                                           const unsigned short* __restrict__ p,
                                           const float* __restrict__ enorm,
                                           int code, float* en) {
#pragma unroll
    for (int ks = 0; ks < 8; ++ks) {
        b[ks].u[0] = *(const uint4*)(p + ks * 32);
        b[ks].u[1] = *(const uint4*)(p + ks * 32 + 16);
    }
    *en = enorm[code];
}

// 8 chained WMMAs over D=256, then distance + running per-row argmin.
__device__ __forceinline__ void tile_step(const Frag32B* __restrict__ a,
                                          const Frag32B* __restrict__ b,
                                          float en, int code,
                                          float* __restrict__ minv,
                                          int* __restrict__ mini) {
    v8f acc = {};
#pragma unroll
    for (int ks = 0; ks < 8; ++ks) {
        acc = __builtin_amdgcn_wmma_f32_16x16x32_bf16(
            false, a[ks].v, false, b[ks].v, (short)0, acc, false, false);
    }
#pragma unroll
    for (int r = 0; r < 8; ++r) {
        const float d = fmaf(-2.0f, acc[r], en);   // ||e||^2 - 2 z.e
        if (d < minv[r]) { minv[r] = d; mini[r] = code; }
    }
}

// ---------------------------------------------------------------------------
// Kernel 3: fused GEMM + argmin, register double-buffered.
// Block = 256 threads = 8 waves; each wave owns 16 rows of z.
// A panel (16 x 256 bf16) lives in 64 VGPRs for the whole kernel; two B-tile
// buffers (64 VGPRs each) alternate so the 16 loads for tile t+1 are in
// flight while the 8 v_wmma_f32_16x16x32_bf16 of tile t execute.
// ---------------------------------------------------------------------------
__global__ __launch_bounds__(256)
void vq_argmin_wmma(const unsigned short* __restrict__ zb,
                    const unsigned short* __restrict__ eb,
                    const float* __restrict__ enorm,
                    int* __restrict__ out) {
    const int lane = threadIdx.x & 31;
    const int wave = threadIdx.x >> 5;
    const int nb   = blockIdx.x * 128 + wave * 16;  // this wave's 16 rows
    const int m    = lane & 15;
    const int hi   = lane >> 4;                     // lane half: 0 or 1

    // ---- Preload A fragments (rows nb..nb+15, all 8 K-steps of 32) -------
    // ISA 16-bit A layout: lane<16 holds K {0..7,16..23}, lane>=16 holds
    // K {8..15,24..31} for row M = lane%16.
    Frag32B a[8];
    {
        const unsigned short* arow = zb + (size_t)(nb + m) * D_DIM;
#pragma unroll
        for (int ks = 0; ks < 8; ++ks) {
            const unsigned short* p = arow + ks * 32 + hi * 8;
            a[ks].u[0] = *(const uint4*)(p);
            a[ks].u[1] = *(const uint4*)(p + 16);
        }
    }

    float minv[8];
    int   mini[8];
#pragma unroll
    for (int r = 0; r < 8; ++r) { minv[r] = 3.4e38f; mini[r] = 0; }

    // Per-lane B column base: lane handles code column cb + m, with K
    // {0..7,16..23} / {8..15,24..31} selected by the lane half (matches A).
    const unsigned short* ebl = eb + (size_t)m * D_DIM + hi * 8;

    Frag32B b0[8], b1[8];
    float   en0, en1;
    load_btile(b0, ebl, enorm, m, &en0);            // tile cb = 0

    // Unrolled x2 so the two register buffers swap roles without copies.
    // Look-ahead index wraps with &(K-1): the final dead prefetch stays
    // in-bounds and branch-free.
    for (int cb = 0; cb < K_CODES; cb += 32) {
        const int nb1 = (cb + 16) & (K_CODES - 1);
        load_btile(b1, ebl + (size_t)nb1 * D_DIM, enorm, nb1 + m, &en1);
        tile_step(a, b0, en0, cb + m, minv, mini);

        const int nb0 = (cb + 32) & (K_CODES - 1);
        load_btile(b0, ebl + (size_t)nb0 * D_DIM, enorm, nb0 + m, &en0);
        tile_step(a, b1, en1, cb + 16 + m, minv, mini);
    }

    // ---- reduce across the 16 lanes of the N dimension -------------------
    // xor masks 1,2,4,8 never touch bit4, so the two row-groups stay apart.
#pragma unroll
    for (int off = 1; off < 16; off <<= 1) {
#pragma unroll
        for (int r = 0; r < 8; ++r) {
            const float ov = __shfl_xor(minv[r], off, 32);
            const int   oi = __shfl_xor(mini[r], off, 32);
            if (ov < minv[r] || (ov == minv[r] && oi < mini[r])) {
                minv[r] = ov; mini[r] = oi;
            }
        }
    }
    if (m == 0) {
#pragma unroll
        for (int r = 0; r < 8; ++r)
            out[nb + hi * 8 + r] = mini[r];
    }
}

// ---------------------------------------------------------------------------
extern "C" void kernel_launch(void* const* d_in, const int* in_sizes, int n_in,
                              void* d_out, int out_size, void* d_ws, size_t ws_size,
                              hipStream_t stream) {
    const float* z   = (const float*)d_in[0];   // [64,256,32,32] fp32
    const float* emb = (const float*)d_in[1];   // [1024,256] fp32

    // workspace layout: z_bf16 (32 MiB) | emb_bf16 (512 KiB) | enorm (4 KiB)
    unsigned short* zb = (unsigned short*)d_ws;
    unsigned short* eb =
        (unsigned short*)((char*)d_ws + (size_t)N_TOTAL * D_DIM * sizeof(unsigned short));
    float* enorm = (float*)((char*)eb + (size_t)K_CODES * D_DIM * sizeof(unsigned short));
    int* out = (int*)d_out;

    dim3 tgrid(64, 32, 8), tblk(32, 8);
    convert_z_nchw_to_nd_bf16<<<tgrid, tblk, 0, stream>>>(z, zb);
    convert_emb_bf16_norms<<<K_CODES, 256, 0, stream>>>(emb, eb, enorm);
    vq_argmin_wmma<<<N_TOTAL / 128, 256, 0, stream>>>(zb, eb, enorm, out);
}
